// hrlcdr_17102559772987
// MI455X (gfx1250) — compile-verified, matrix operand order checked by reading, not run on hardware
//
#include <hip/hip_runtime.h>
#include <hip/hip_bf16.h>

// ---------------------------------------------------------------------------
// Types for CDNA5 WMMA (wave32, v_wmma_f32_16x16x32_bf16)
// ---------------------------------------------------------------------------
typedef __bf16          v16bf  __attribute__((ext_vector_type(16)));
typedef float           v8f    __attribute__((ext_vector_type(8)));
typedef unsigned short  u16x16 __attribute__((ext_vector_type(16)));
typedef float           f32x4  __attribute__((ext_vector_type(4), aligned(4)));

#define GAMMA_C 8.7f
#define BN_EPS_C 1e-5f

static_assert(sizeof(__hip_bfloat16) == 2, "bf16 must be 2 bytes");

// RNE f32 -> bf16 bits via the HIP/hardware conversion path
__device__ __forceinline__ unsigned short f2bf(float f) {
  __hip_bfloat16 h = __float2bfloat16(f);
  return __builtin_bit_cast(unsigned short, h);
}

// ---------------------------------------------------------------------------
// One K=32 step of the WMMA GEMM.
//   NG: B column guard needed (n < Nb)    -- only last-N wave
//   KG: B k guard needed (k < Kw)         -- only the single tail step
// Wave tile: 32(M) x 64(N); 2 A fragments x 4 B fragments -> 8 WMMAs.
// ---------------------------------------------------------------------------
template <bool NG, bool KG>
__device__ __forceinline__ void gemm_step(
    const unsigned short* __restrict__ Ap0,
    const unsigned short* __restrict__ Ap1, int kk,
    const float* __restrict__ Bp, int tileN, int lr, int lh,
    int Kw, int ldb, int Nb, v8f (&acc)[2][4])
{
  union { u16x16 u; v16bf b; } a0, a1;
  a0.u = *(const u16x16*)(Ap0 + kk);          // rows tileM+lr   (zero-padded)
  a1.u = *(const u16x16*)(Ap1 + kk);          // rows tileM+16+lr
  const int k0 = kk + lh * 16;

  v16bf bfrag[4];
#pragma unroll
  for (int j = 0; j < 4; ++j) {
    const int n = tileN + j * 16 + lr;
    union { u16x16 u; v16bf b; } bc;
    if (!NG && !KG) {
      const float* bp = Bp + (size_t)n * ldb + k0;
#pragma unroll
      for (int t = 0; t < 16; t += 4) {
        f32x4 fv = *(const f32x4*)(bp + t);
        bc.u[t + 0] = f2bf(fv[0]);
        bc.u[t + 1] = f2bf(fv[1]);
        bc.u[t + 2] = f2bf(fv[2]);
        bc.u[t + 3] = f2bf(fv[3]);
      }
    } else {
#pragma unroll
      for (int t = 0; t < 16; ++t) {
        bool ok = (!NG || n < Nb) && (!KG || (k0 + t) < Kw);
        float x = ok ? Bp[(size_t)n * ldb + k0 + t] : 0.0f;
        bc.u[t] = f2bf(x);
      }
    }
    bfrag[j] = bc.b;
  }

#pragma unroll
  for (int j = 0; j < 4; ++j) {
    acc[0][j] = __builtin_amdgcn_wmma_f32_16x16x32_bf16(
        false, a0.b, false, bfrag[j], (short)0, acc[0][j], false, false);
    acc[1][j] = __builtin_amdgcn_wmma_f32_16x16x32_bf16(
        false, a1.b, false, bfrag[j], (short)0, acc[1][j], false, false);
  }
}

// ---------------------------------------------------------------------------
// Generic WMMA GEMM:  C[M,N] = A_bf16[M,Ktot] * B_f32[N,Kw]^T  (TN form)
//   A: zero-padded bf16 (rows padded to 32, K padded to 32-multiples)
//   B: up to 4 f32 weight slices, each [N, Kw] row-major (K contiguous);
//      slice s pairs with A columns [s*sliceKpad, s*sliceKpad+Kw).
//   Wave = 32x64 strip; block = 8 waves = 32(M) x 512(N).
// Epilogues:
//   MODE 0: C = acc + bias[n]
//   MODE 1: C = relu((acc + bias[n]) * (base[m,n]+1))
//   MODE 2: C = sigmoid(GAMMA * acc * rsqrt(u[m]*v[n]))
// ---------------------------------------------------------------------------
template <int MODE>
__global__ __launch_bounds__(256) void hrl_wmma_gemm(
    const unsigned short* __restrict__ A, int lda,
    const float* __restrict__ B0, const float* __restrict__ B1,
    const float* __restrict__ B2, const float* __restrict__ B3,
    int nslices, int sliceKpad, int Kw, int ldb,
    const float* __restrict__ bias,
    const float* __restrict__ base,
    const float* __restrict__ uvec, const float* __restrict__ vvec,
    float* __restrict__ C, int ldc,
    int Mstore, int Nb, int Nstore)
{
  const int lane  = threadIdx.x & 31;
  const int wave  = threadIdx.x >> 5;
  const int tileM = blockIdx.y * 32;
  const int tileN = blockIdx.x * 512 + wave * 64;
  const int lr = lane & 15;  // A row-in-tile / B col-in-tile
  const int lh = lane >> 4;  // K-half selector

  v8f acc[2][4] = {};

  const bool nfull = (tileN + 64) <= Nb;   // wave-uniform branch
  const int  kmain = Kw & ~31;

  for (int s = 0; s < nslices; ++s) {
    const float* Bp = (s == 0) ? B0 : (s == 1) ? B1 : (s == 2) ? B2 : B3;
    const unsigned short* Ap0 =
        A + (size_t)(tileM + lr) * lda + (size_t)s * sliceKpad + lh * 16;
    const unsigned short* Ap1 = Ap0 + (size_t)16 * lda;

    if (nfull) {
      for (int kk = 0; kk < kmain; kk += 32)
        gemm_step<false, false>(Ap0, Ap1, kk, Bp, tileN, lr, lh, Kw, ldb, Nb, acc);
      if (kmain < sliceKpad)
        gemm_step<false, true>(Ap0, Ap1, kmain, Bp, tileN, lr, lh, Kw, ldb, Nb, acc);
    } else {
      for (int kk = 0; kk < kmain; kk += 32)
        gemm_step<true, false>(Ap0, Ap1, kk, Bp, tileN, lr, lh, Kw, ldb, Nb, acc);
      if (kmain < sliceKpad)
        gemm_step<true, true>(Ap0, Ap1, kmain, Bp, tileN, lr, lh, Kw, ldb, Nb, acc);
    }
  }

  // Epilogue. C VGPR layout: vgpr r -> M = r + 8*lh, N = lane&15.
#pragma unroll
  for (int mi = 0; mi < 2; ++mi) {
#pragma unroll
    for (int j = 0; j < 4; ++j) {
#pragma unroll
      for (int r = 0; r < 8; ++r) {
        const int row = tileM + mi * 16 + r + lh * 8;
        const int col = tileN + j * 16 + lr;
        if (row < Mstore && col < Nstore) {
          float a = acc[mi][j][r];
          if (MODE == 0) {
            C[(size_t)row * ldc + col] = a + bias[col];
          } else if (MODE == 1) {
            float h = a + bias[col];
            float y = h * (base[(size_t)row * ldc + col] + 1.0f);
            C[(size_t)row * ldc + col] = fmaxf(y, 0.0f);
          } else {
            float t = GAMMA_C * a * rsqrtf(uvec[row] * vvec[col]);
            C[(size_t)row * ldc + col] = 1.0f / (1.0f + __expf(-t));
          }
        }
      }
    }
  }
}

// ---------------------------------------------------------------------------
// Small helper kernels
// ---------------------------------------------------------------------------
__global__ void hrl_zero_u16(unsigned short* __restrict__ p, int n) {
  int i = blockIdx.x * 256 + threadIdx.x;
  if (i < n) p[i] = 0;
}

// rs = sum_d adj[c,:] + 1 ; dx = rs^-1/2 ; dc = 1/rs + 1
__global__ void hrl_adj_row(const float* __restrict__ adj, int M, int N,
                            float* __restrict__ dx, float* __restrict__ dc) {
  int c = blockIdx.x * blockDim.x + threadIdx.x;
  if (c >= M) return;
  float s = 1.0f;
  const float* r = adj + (size_t)c * N;
  for (int d = 0; d < N; ++d) s += r[d];
  dx[c] = rsqrtf(s);
  dc[c] = 1.0f / s + 1.0f;
}

// cs = sum_c adj[:,d] + 1 ; dy = cs^-1/2 ; dd = 1/cs + 1
__global__ void hrl_adj_col(const float* __restrict__ adj, int M, int N,
                            float* __restrict__ dy, float* __restrict__ dd) {
  int d = blockIdx.x * blockDim.x + threadIdx.x;
  if (d >= N) return;
  float s = 1.0f;
  for (int c = 0; c < M; ++c) s += adj[(size_t)c * N + d];
  dy[d] = rsqrtf(s);
  dd[d] = 1.0f / s + 1.0f;
}

// out[i] = 1 / (rowsum(H) + 1)
__global__ void hrl_hyper_inv(const float* __restrict__ H, int n,
                              float* __restrict__ out) {
  int i = blockIdx.x * blockDim.x + threadIdx.x;
  if (i >= n) return;
  float s = 1.0f;
  const float* r = H + (size_t)i * n;
  for (int j = 0; j < n; ++j) s += r[j];
  out[i] = 1.0f / s;
}

// Row z-score (ddof=1) of cell_exprs -> zero-padded bf16 [rows, ldd]
__global__ __launch_bounds__(256) void hrl_znorm(
    const float* __restrict__ x, int K,
    unsigned short* __restrict__ dst, int ldd) {
  __shared__ float ssum[256];
  __shared__ float ssq[256];
  const int row = blockIdx.x, tid = threadIdx.x;
  const float* xr = x + (size_t)row * K;
  float s = 0.f, q = 0.f;
  for (int k = tid; k < K; k += 256) { float v = xr[k]; s += v; q += v * v; }
  ssum[tid] = s; ssq[tid] = q; __syncthreads();
  for (int off = 128; off; off >>= 1) {
    if (tid < off) { ssum[tid] += ssum[tid + off]; ssq[tid] += ssq[tid + off]; }
    __syncthreads();
  }
  float mean = ssum[0] / K;
  float var  = (ssq[0] - K * mean * mean) / (K - 1);
  float inv  = rsqrtf(var);
  for (int k = tid; k < K; k += 256)
    dst[(size_t)row * ldd + k] = f2bf((xr[k] - mean) * inv);
}

// f32 [M,K] -> zero-padded bf16 [M, ldd]
__global__ void hrl_cvt_pad(const float* __restrict__ src, int M, int K,
                            unsigned short* __restrict__ dst, int ldd) {
  int i = blockIdx.x * 256 + threadIdx.x;
  if (i >= M * K) return;
  int m = i / K, k = i - m * K;
  dst[(size_t)m * ldd + k] = f2bf(src[i]);
}

// BatchNorm1d training stats over rows (biased var), per column n
__global__ void hrl_bn(const float* __restrict__ pre, int rows, int N,
                       const float* __restrict__ g, const float* __restrict__ b,
                       float* __restrict__ out) {
  int n = blockIdx.x * blockDim.x + threadIdx.x;
  if (n >= N) return;
  float s = 0.f, q = 0.f;
  for (int r = 0; r < rows; ++r) {
    float v = pre[(size_t)r * N + n]; s += v; q += v * v;
  }
  float mu = s / rows;
  float var = q / rows - mu * mu;
  float sc = rsqrtf(var + BN_EPS_C) * g[n];
  float sh = b[n] - mu * sc;
  for (int r = 0; r < rows; ++r)
    out[(size_t)r * N + n] = pre[(size_t)r * N + n] * sc + sh;
}

// P1[c,d] = dx[c]*adj[c,d]*dy[d] ; P3[d,c] = dy[d]*adj[c,d]*dx[c]
__global__ void hrl_p1p3(const float* __restrict__ adj, int M, int N,
                         const float* __restrict__ dx, const float* __restrict__ dy,
                         float* __restrict__ P1, float* __restrict__ P3) {
  int i = blockIdx.x * 256 + threadIdx.x;
  if (i >= M * N) return;
  int c = i / N, d = i - c * N;
  float v = adj[i] * dx[c] * dy[d];
  P1[(size_t)c * N + d] = v;
  P3[(size_t)d * M + c] = v;
}

// P[i,j] = delta(i,j) - hinv[i]*H[i,j]*hinv[j]
__global__ void hrl_chaotu(const float* __restrict__ H,
                           const float* __restrict__ hinv, int n,
                           float* __restrict__ P) {
  int i = blockIdx.x * 256 + threadIdx.x;
  if (i >= n * n) return;
  int r = i / n, c = i - r * n;
  P[i] = ((r == c) ? 1.0f : 0.0f) - hinv[r] * H[i] * hinv[c];
}

// bsum_cell = b0+b1+b4+b6 ; bsum_drug = b2+b3+b5+b7
__global__ void hrl_bias4(const float* __restrict__ b, int N,
                          float* __restrict__ bc, float* __restrict__ bd) {
  int n = blockIdx.x * blockDim.x + threadIdx.x;
  if (n >= N) return;
  bc[n] = b[0 * N + n] + b[1 * N + n] + b[4 * N + n] + b[6 * N + n];
  bd[n] = b[2 * N + n] + b[3 * N + n] + b[5 * N + n] + b[7 * N + n];
}

// dst[m, colOff+n] = bf16(scale[m] * src[m,n])   (diag-scaled cat slice)
__global__ void hrl_scale_cat(const float* __restrict__ src,
                              const float* __restrict__ scale, int M, int N,
                              unsigned short* __restrict__ dst, int ldd, int colOff) {
  int i = blockIdx.x * 256 + threadIdx.x;
  if (i >= M * N) return;
  int m = i / N, n = i - m * N;
  dst[(size_t)m * ldd + colOff + n] = f2bf(scale[m] * src[i]);
}

// C[m, colOff+n] = bf16( sum_k P[m,k] * Q[k,n] )   small-K mixing GEMM (f32)
__global__ __launch_bounds__(256) void hrl_mixer(
    const float* __restrict__ P, int ldp,
    const float* __restrict__ Q, int ldq,
    int K, int N,
    unsigned short* __restrict__ dst, int ldd, int colOff) {
  int n = blockIdx.x * 256 + threadIdx.x;
  int m = blockIdx.y;
  if (n >= N) return;
  const float* p = P + (size_t)m * ldp;
  float acc = 0.f;
#pragma unroll 4
  for (int k = 0; k < K; ++k) acc = fmaf(p[k], Q[(size_t)k * ldq + n], acc);
  dst[(size_t)m * ldd + colOff + n] = f2bf(acc);
}

// Per-row: mean, centered values -> bf16 padded, and norm = sum(c^2)
__global__ __launch_bounds__(256) void hrl_rowstat_bf(
    const float* __restrict__ x, int N,
    unsigned short* __restrict__ xc, int ldx, float* __restrict__ norm) {
  __shared__ float sbuf[256];
  const int row = blockIdx.x, tid = threadIdx.x;
  const float* xr = x + (size_t)row * N;
  float s = 0.f;
  for (int k = tid; k < N; k += 256) s += xr[k];
  sbuf[tid] = s; __syncthreads();
  for (int off = 128; off; off >>= 1) {
    if (tid < off) sbuf[tid] += sbuf[tid + off];
    __syncthreads();
  }
  float mean = sbuf[0] / N;
  __syncthreads();
  float q = 0.f;
  for (int k = tid; k < N; k += 256) {
    float c = xr[k] - mean;
    q += c * c;
    xc[(size_t)row * ldx + k] = f2bf(c);
  }
  sbuf[tid] = q; __syncthreads();
  for (int off = 128; off; off >>= 1) {
    if (tid < off) sbuf[tid] += sbuf[tid + off];
    __syncthreads();
  }
  if (tid == 0) norm[row] = sbuf[0];
}

// Same but centered values stored f32 (B-operand of the corr WMMA GEMM)
__global__ __launch_bounds__(256) void hrl_rowstat_f32(
    const float* __restrict__ x, int N,
    float* __restrict__ yc, float* __restrict__ norm) {
  __shared__ float sbuf[256];
  const int row = blockIdx.x, tid = threadIdx.x;
  const float* xr = x + (size_t)row * N;
  float s = 0.f;
  for (int k = tid; k < N; k += 256) s += xr[k];
  sbuf[tid] = s; __syncthreads();
  for (int off = 128; off; off >>= 1) {
    if (tid < off) sbuf[tid] += sbuf[tid + off];
    __syncthreads();
  }
  float mean = sbuf[0] / N;
  __syncthreads();
  float q = 0.f;
  for (int k = tid; k < N; k += 256) {
    float c = xr[k] - mean;
    q += c * c;
    yc[(size_t)row * N + k] = c;
  }
  sbuf[tid] = q; __syncthreads();
  for (int off = 128; off; off >>= 1) {
    if (tid < off) sbuf[tid] += sbuf[tid + off];
    __syncthreads();
  }
  if (tid == 0) norm[row] = sbuf[0];
}

// ---------------------------------------------------------------------------
// Host-side orchestration
// ---------------------------------------------------------------------------
extern "C" void kernel_launch(void* const* d_in, const int* in_sizes, int n_in,
                              void* d_out, int out_size, void* d_ws, size_t ws_size,
                              hipStream_t stream) {
  (void)in_sizes; (void)n_in; (void)out_size; (void)ws_size;

  constexpr int NC = 60, ND = 952, D = 2040, NG = 17737, NFP = 881;
  constexpr int KGp = 17760;          // NG padded to 32
  constexpr int KFp = 896;            // NFP padded to 32
  constexpr int Dp  = 2048;           // D padded to 32
  constexpr int MCp = 64, MDp = 960;  // M padded to 32
  constexpr int KCAT = 4 * Dp;        // 8192: fused 4-GEMM K

  const float* adj         = (const float*)d_in[0];
  const float* cell_exprs  = (const float*)d_in[1];
  const float* drug_finger = (const float*)d_in[2];
  const float* cell_hyper  = (const float*)d_in[3];
  const float* drug_hyper  = (const float*)d_in[4];
  const float* W_agg       = (const float*)d_in[5];
  const float* b_agg       = (const float*)d_in[6];
  const float* ldd_w       = (const float*)d_in[7];
  const float* ldd_b       = (const float*)d_in[8];
  const float* lcc_w       = (const float*)d_in[9];
  const float* lcc_b       = (const float*)d_in[10];
  const float* bnd_g       = (const float*)d_in[11];
  const float* bnd_b       = (const float*)d_in[12];
  const float* bnc_g       = (const float*)d_in[13];
  const float* bnc_b       = (const float*)d_in[14];
  float* out = (float*)d_out;

  // ---- workspace layout (256B-aligned bump allocator)
  char* wsp = (char*)d_ws;
  auto alloc = [&](size_t bytes) -> char* {
    char* p = wsp;
    wsp += (bytes + 255) & ~(size_t)255;
    return p;
  };
  float* dx  = (float*)alloc(NC * 4);
  float* dc  = (float*)alloc(NC * 4);
  float* dy  = (float*)alloc(ND * 4);
  float* dd  = (float*)alloc(ND * 4);
  float* b4h = (float*)alloc(NC * 4);
  float* a4h = (float*)alloc(ND * 4);
  float* bc  = (float*)alloc(D * 4);
  float* bd  = (float*)alloc(D * 4);
  float* lxx = (float*)alloc(NC * 4);
  float* lyy = (float*)alloc(ND * 4);
  unsigned short* cellfeat_bf = (unsigned short*)alloc((size_t)MCp * KGp * 2);
  unsigned short* drugfp_bf   = (unsigned short*)alloc((size_t)MDp * KFp * 2);
  unsigned short* Xcat_cell   = (unsigned short*)alloc((size_t)MCp * KCAT * 2);
  unsigned short* Xcat_drug   = (unsigned short*)alloc((size_t)MDp * KCAT * 2);
  unsigned short* xc_bf       = (unsigned short*)alloc((size_t)MCp * Dp * 2);
  float* exp_pre = (float*)alloc((size_t)NC * D * 4);
  float* fig_pre = (float*)alloc((size_t)ND * D * 4);
  float* expf    = (float*)alloc((size_t)NC * D * 4);
  float* figf    = (float*)alloc((size_t)ND * D * 4);
  float* cellagg = (float*)alloc((size_t)NC * D * 4);
  float* drugagg = (float*)alloc((size_t)ND * D * 4);
  float* yc      = (float*)alloc((size_t)ND * D * 4);
  float* P1      = (float*)alloc((size_t)NC * ND * 4);
  float* P3      = (float*)alloc((size_t)ND * NC * 4);
  float* P6      = (float*)alloc((size_t)NC * NC * 4);
  float* P7      = (float*)alloc((size_t)ND * ND * 4);

  auto blocks = [](long n) { return (unsigned)((n + 255) / 256); };

  // ---- 0) zero the padded bf16 staging buffers (deterministic each call)
  hrl_zero_u16<<<blocks((long)MCp * KGp), 256, 0, stream>>>(cellfeat_bf, MCp * KGp);
  hrl_zero_u16<<<blocks((long)MDp * KFp), 256, 0, stream>>>(drugfp_bf, MDp * KFp);
  hrl_zero_u16<<<blocks((long)MCp * KCAT), 256, 0, stream>>>(Xcat_cell, MCp * KCAT);
  hrl_zero_u16<<<blocks((long)MDp * KCAT), 256, 0, stream>>>(Xcat_drug, MDp * KCAT);
  hrl_zero_u16<<<blocks((long)MCp * Dp), 256, 0, stream>>>(xc_bf, MCp * Dp);

  // ---- 1) adjacency / hypergraph scalings
  hrl_adj_row<<<1, 64, 0, stream>>>(adj, NC, ND, dx, dc);
  hrl_adj_col<<<blocks(ND), 256, 0, stream>>>(adj, NC, ND, dy, dd);
  hrl_hyper_inv<<<1, 64, 0, stream>>>(cell_hyper, NC, b4h);
  hrl_hyper_inv<<<blocks(ND), 256, 0, stream>>>(drug_hyper, ND, a4h);

  // ---- 2) feature prep: z-norm rows (bf16 A), drug fingerprints (bf16 A)
  hrl_znorm<<<NC, 256, 0, stream>>>(cell_exprs, NG, cellfeat_bf, KGp);
  hrl_cvt_pad<<<blocks((long)ND * NFP), 256, 0, stream>>>(drug_finger, ND, NFP,
                                                          drugfp_bf, KFp);

  // ---- 3) input projections (WMMA, f32 weights cvt in-register), then BN
  hrl_wmma_gemm<0><<<dim3(4, MCp / 32), 256, 0, stream>>>(
      cellfeat_bf, KGp, lcc_w, nullptr, nullptr, nullptr,
      1, KGp, NG, NG, lcc_b, nullptr, nullptr, nullptr,
      exp_pre, D, NC, D, D);
  hrl_wmma_gemm<0><<<dim3(4, MDp / 32), 256, 0, stream>>>(
      drugfp_bf, KFp, ldd_w, nullptr, nullptr, nullptr,
      1, KFp, NFP, NFP, ldd_b, nullptr, nullptr, nullptr,
      fig_pre, D, ND, D, D);
  hrl_bn<<<blocks(D), 256, 0, stream>>>(exp_pre, NC, D, bnc_g, bnc_b, expf);
  hrl_bn<<<blocks(D), 256, 0, stream>>>(fig_pre, ND, D, bnd_g, bnd_b, figf);

  // ---- 4) mixing operators
  hrl_p1p3<<<blocks((long)NC * ND), 256, 0, stream>>>(adj, NC, ND, dx, dy, P1, P3);
  hrl_chaotu<<<blocks((long)NC * NC), 256, 0, stream>>>(cell_hyper, b4h, NC, P6);
  hrl_chaotu<<<blocks((long)ND * ND), 256, 0, stream>>>(drug_hyper, a4h, ND, P7);

  // ---- 5) build concatenated bf16 inputs Xcat (slice order = weight order)
  hrl_scale_cat<<<blocks((long)NC * D), 256, 0, stream>>>(expf, dc, NC, D,
                                                          Xcat_cell, KCAT, 0 * Dp);
  hrl_mixer<<<dim3(8, NC), 256, 0, stream>>>(P1, ND, figf, D, ND, D,
                                             Xcat_cell, KCAT, 1 * Dp);
  hrl_mixer<<<dim3(8, NC), 256, 0, stream>>>(cell_hyper, NC, expf, D, NC, D,
                                             Xcat_cell, KCAT, 2 * Dp);
  hrl_mixer<<<dim3(8, NC), 256, 0, stream>>>(P6, NC, expf, D, NC, D,
                                             Xcat_cell, KCAT, 3 * Dp);
  hrl_scale_cat<<<blocks((long)ND * D), 256, 0, stream>>>(figf, dd, ND, D,
                                                          Xcat_drug, KCAT, 0 * Dp);
  hrl_mixer<<<dim3(8, ND), 256, 0, stream>>>(P3, NC, expf, D, NC, D,
                                             Xcat_drug, KCAT, 1 * Dp);
  hrl_mixer<<<dim3(8, ND), 256, 0, stream>>>(drug_hyper, ND, figf, D, ND, D,
                                             Xcat_drug, KCAT, 2 * Dp);
  hrl_mixer<<<dim3(8, ND), 256, 0, stream>>>(P7, ND, figf, D, ND, D,
                                             Xcat_drug, KCAT, 3 * Dp);

  hrl_bias4<<<blocks(D), 256, 0, stream>>>(b_agg, D, bc, bd);

  // ---- 6) fused 4-way aggregation GEMMs (K = 8192) with bias*(base+1)+relu
  const size_t WS = (size_t)D * D;
  hrl_wmma_gemm<1><<<dim3(4, MCp / 32), 256, 0, stream>>>(
      Xcat_cell, KCAT, W_agg + 0 * WS, W_agg + 1 * WS, W_agg + 4 * WS, W_agg + 6 * WS,
      4, Dp, D, D, bc, expf, nullptr, nullptr,
      cellagg, D, NC, D, D);
  hrl_wmma_gemm<1><<<dim3(4, MDp / 32), 256, 0, stream>>>(
      Xcat_drug, KCAT, W_agg + 2 * WS, W_agg + 3 * WS, W_agg + 5 * WS, W_agg + 7 * WS,
      4, Dp, D, D, bd, figf, nullptr, nullptr,
      drugagg, D, ND, D, D);

  // ---- 7) correlation decoder: center rows, WMMA lxy, fused sigmoid
  hrl_rowstat_bf<<<NC, 256, 0, stream>>>(cellagg, D, xc_bf, Dp, lxx);
  hrl_rowstat_f32<<<ND, 256, 0, stream>>>(drugagg, D, yc, lyy);
  hrl_wmma_gemm<2><<<dim3(2, MCp / 32), 256, 0, stream>>>(
      xc_bf, Dp, yc, nullptr, nullptr, nullptr,
      1, Dp, D, D, nullptr, nullptr, lxx, lyy,
      out, ND, NC, ND, ND);
}
